// Gemma2TransformerBlock_25555055411771
// MI455X (gfx1250) — matrix operand order repaired and checked by reading, not compiled, not run on hardware
//
#include <hip/hip_runtime.h>
#include <math.h>

// ---------------------------------------------------------------------------
// Gemma2 transformer block for MI455X (gfx1250).
// bf16 WMMA for all matrix math; async global->LDS staging (ASYNCcnt path);
// weights pre-transposed once so all LDS staging is contiguous b128 traffic.
// ---------------------------------------------------------------------------

typedef __bf16 bf16_t;
typedef __attribute__((ext_vector_type(16))) __bf16 v16bf;
typedef __attribute__((ext_vector_type(8)))  __bf16 v8bf;
typedef __attribute__((ext_vector_type(8)))  float  v8f;

static constexpr int B_   = 2;
static constexpr int S_   = 2048;
static constexpr int H_   = 2304;
static constexpr int NH_  = 8;
static constexpr int NKV_ = 4;
static constexpr int HD_  = 256;
static constexpr int FF_  = 9216;
static constexpr int M_   = B_ * S_;      // 4096 token rows

#define WMMA_BF16(a, b, c) \
  __builtin_amdgcn_wmma_f32_16x16x32_bf16(false, (a), false, (b), (short)0, (c), false, false)

__device__ __forceinline__ v16bf cat8(v8bf lo, v8bf hi) {
  return __builtin_shufflevector(lo, hi, 0,1,2,3,4,5,6,7,8,9,10,11,12,13,14,15);
}

// A-matrix fragment (16x32 bf16): lane<16 holds K {0..15}, lane>=16 K {16..31}
__device__ __forceinline__ v16bf a_frag(const bf16_t* base, int r, int half, int stride) {
  const bf16_t* p = base + r * stride;
  v8bf lo = *(const v8bf*)(p + half * 16);
  v8bf hi = *(const v8bf*)(p + half * 16 + 8);
  return cat8(lo, hi);
}

// ---- CDNA5 async global->LDS copy (16B per lane), tracked by ASYNCcnt ----
__device__ __forceinline__ void async_b128(const bf16_t* gsrc, bf16_t* ldst) {
  unsigned lds = (unsigned)(uintptr_t)ldst;          // low 32 bits = LDS offset
  unsigned long long ga = (unsigned long long)(uintptr_t)gsrc;
  asm volatile("global_load_async_to_lds_b128 %0, %1, off"
               :: "v"(lds), "v"(ga) : "memory");
}
#define ASYNC_WAIT(n) asm volatile("s_wait_asynccnt %0" :: "n"(n) : "memory")

// ---------------------------------------------------------------------------
// One-time weight prep: f32 [K][N] -> bf16 [N][K] (transpose via LDS tile)
// ---------------------------------------------------------------------------
__global__ void k_cast_transpose(const float* __restrict__ src, bf16_t* __restrict__ dst,
                                 int K, int N) {
  __shared__ float t[64][65];
  const int tid = threadIdx.x;
  const long k0 = (long)blockIdx.y * 64, n0 = (long)blockIdx.x * 64;
#pragma unroll
  for (int it = 0; it < 16; it++) {
    int e = it * 256 + tid; int rr = e >> 6, cc = e & 63;
    t[rr][cc] = src[(k0 + rr) * N + n0 + cc];
  }
  __syncthreads();
#pragma unroll
  for (int it = 0; it < 16; it++) {
    int e = it * 256 + tid; int rr = e >> 6, cc = e & 63;
    dst[(n0 + rr) * (long)K + k0 + cc] = (bf16_t)t[cc][rr];
  }
}

// ---------------------------------------------------------------------------
// RMSNorm: out = (optional res +) x * rsqrt(mean(x^2)+eps) * (1+g)
// ---------------------------------------------------------------------------
__global__ void k_rmsnorm(const float* __restrict__ in, const float* __restrict__ g,
                          const float* __restrict__ res, float* __restrict__ outf,
                          bf16_t* __restrict__ outb, int cols) {
  const long row = blockIdx.x;
  const float* r0 = in + row * cols;
  __shared__ float red[256];
  float s = 0.f;
  for (int c = threadIdx.x; c < cols; c += 256) { float v = r0[c]; s += v * v; }
  red[threadIdx.x] = s;
  __syncthreads();
  for (int w = 128; w > 0; w >>= 1) {
    if ((int)threadIdx.x < w) red[threadIdx.x] += red[threadIdx.x + w];
    __syncthreads();
  }
  const float rs = rsqrtf(red[0] / (float)cols + 1e-6f);
  for (int c = threadIdx.x; c < cols; c += 256) {
    float v = r0[c] * rs * (1.0f + g[c]);
    if (res)  v += res[row * cols + c];
    if (outf) outf[row * cols + c] = v;
    if (outb) outb[row * cols + c] = (bf16_t)v;
  }
}

// ---------------------------------------------------------------------------
// bf16 GEMM: C[M,N] = A[M,K] @ W[K,N] with Wt = W^T ([N][K], pre-transposed).
// Block 256 threads (8 waves), 128x128 block tile, wave tile 32x64, K-step 64.
// Double-buffered LDS, async global->LDS staging, uniform counted-wait loop
// (always prefetch; last iteration wraps to k=0 and is simply never consumed).
// ---------------------------------------------------------------------------
__global__ __launch_bounds__(256, 2)
void k_gemm_bf16(const bf16_t* __restrict__ A, const bf16_t* __restrict__ Wt,
                 float* __restrict__ C, int M, int N, int K) {
  __shared__ __attribute__((aligned(32))) bf16_t Al[2][128 * 64];
  __shared__ __attribute__((aligned(32))) bf16_t Bl[2][128 * 64];   // [col][k]
  const int tid  = threadIdx.x;
  const int lane = tid & 31, wave = tid >> 5;
  const int half = lane >> 4, r = lane & 15;
  const int wm = wave >> 1, wn = wave & 1;      // wave tile rows wm*32, cols wn*64
  const long brow = (long)blockIdx.y * 128;
  const long bcol = (long)blockIdx.x * 128;

  v8f acc[2][4];
#pragma unroll
  for (int i = 0; i < 2; i++)
#pragma unroll
    for (int j = 0; j < 4; j++) acc[i][j] = (v8f){};

  // staging geometry: element e = it*2048 + tid*8 -> row (tid>>3)+it*32, col (tid&7)*8
  const bf16_t* agp = A  + (brow + (tid >> 3)) * (long)K + (tid & 7) * 8;
  const bf16_t* bgp = Wt + (bcol + (tid >> 3)) * (long)K + (tid & 7) * 8;
  const int soff = (tid >> 3) * 64 + (tid & 7) * 8;

  auto issue = [&](int k0, int buf) {
    bf16_t* al = &Al[buf][soff];
    bf16_t* bl = &Bl[buf][soff];
#pragma unroll
    for (int it = 0; it < 4; it++)
      async_b128(agp + (long)it * 32 * K + k0, al + it * 32 * 64);
#pragma unroll
    for (int it = 0; it < 4; it++)
      async_b128(bgp + (long)it * 32 * K + k0, bl + it * 32 * 64);
  };

  issue(0, 0);
  int cur = 0;
  for (int k0 = 0; k0 < K; k0 += 64) {
    const int knext = (k0 + 64 < K) ? k0 + 64 : 0;  // uniform wrap prefetch
    issue(knext, cur ^ 1);
    ASYNC_WAIT(8);            // release exactly the 8 loads of the current tile
    __syncthreads();
    const bf16_t* Ab = Al[cur];
    const bf16_t* Bb = Bl[cur];
#pragma unroll
    for (int ks = 0; ks < 64; ks += 32) {
      v16bf a0 = a_frag(Ab + (wm * 32 + 0)  * 64 + ks, r, half, 64);
      v16bf a1 = a_frag(Ab + (wm * 32 + 16) * 64 + ks, r, half, 64);
#pragma unroll
      for (int t = 0; t < 4; t++) {
        v16bf bt = *(const v16bf*)&Bb[(wn * 64 + t * 16 + r) * 64 + ks + half * 16];
        acc[0][t] = WMMA_BF16(a0, bt, acc[0][t]);
        acc[1][t] = WMMA_BF16(a1, bt, acc[1][t]);
      }
    }
    __syncthreads();
    cur ^= 1;
  }
  ASYNC_WAIT(0);  // drain the wrapped prefetch before LDS goes away

  const long crow = brow + wm * 32;
  const long ccol = bcol + wn * 64;
#pragma unroll
  for (int ti = 0; ti < 2; ti++)
#pragma unroll
    for (int t = 0; t < 4; t++)
#pragma unroll
      for (int i = 0; i < 8; i++)
        C[(crow + ti * 16 + i + 8 * half) * (long)N + ccol + t * 16 + r] = acc[ti][t][i];
}

// ---------------------------------------------------------------------------
// RoPE: f32 [B,S,nh,HD] -> bf16 [B,nh,S,HD]
// ---------------------------------------------------------------------------
__global__ void k_rope(const float* __restrict__ src, bf16_t* __restrict__ dst, int nh) {
  long idx = (long)blockIdx.x * blockDim.x + threadIdx.x;
  const long total = (long)B_ * S_ * nh * (HD_ / 2);
  if (idx >= total) return;
  const int i = (int)(idx & 127);
  long t = idx >> 7;
  const int h = (int)(t % nh); t /= nh;
  const int s = (int)(t % S_);
  const int b = (int)(t / S_);
  const float inv = expf(-(float)(2 * i) / (float)HD_ * logf(10000.0f));
  const float ang = (float)s * inv;
  const float c = cosf(ang), sn = sinf(ang);
  const long sb = (((long)b * S_ + s) * nh + h) * HD_;
  const float x0 = src[sb + i], x1 = src[sb + i + 128];
  const long db = (((long)b * nh + h) * S_ + s) * HD_;
  dst[db + i]       = (bf16_t)(x0 * c - x1 * sn);
  dst[db + i + 128] = (bf16_t)(x1 * c + x0 * sn);
}

// V: f32 [B,S,NKV,HD] -> bf16 [B,NKV,HD,S]  (pre-transposed for attention P.V)
__global__ void k_vperm(const float* __restrict__ src, bf16_t* __restrict__ dst) {
  long idx = (long)blockIdx.x * blockDim.x + threadIdx.x;
  const long total = (long)B_ * NKV_ * HD_ * S_;
  if (idx >= total) return;
  const int s = (int)(idx & (S_ - 1));
  long t = idx >> 11;                 // S_=2048
  const int d = (int)(t & (HD_ - 1));
  t >>= 8;                            // HD_=256
  const int h = (int)(t % NKV_);
  const int b = (int)(t / NKV_);
  dst[idx] = (bf16_t)src[(((long)b * S_ + s) * NKV_ + h) * HD_ + d];
}

// ---------------------------------------------------------------------------
// Flash attention with WMMA (QK^T and P.V), softcap + causal + online softmax.
// Grid: (S/64, NH, B). Block 256 = 8 waves: qsub=w&3 (16 q rows), hh=w>>2.
// Q,K: bf16 [B,heads,S,HD]; V: bf16 [B,NKV,HD,S] (transposed).
// O written bf16 [B,S,NH*HD] (A-matrix of the O-projection GEMM).
// ---------------------------------------------------------------------------
__global__ __launch_bounds__(256, 1)
void k_attn(const bf16_t* __restrict__ Q, const bf16_t* __restrict__ K,
            const bf16_t* __restrict__ V, bf16_t* __restrict__ O) {
  constexpr int QT = 64, KT = 32;
  __shared__ __attribute__((aligned(32))) bf16_t Qs[QT * HD_];
  __shared__ __attribute__((aligned(32))) bf16_t Ks[KT * HD_];
  __shared__ __attribute__((aligned(32))) bf16_t Vt[HD_ * KT];   // [hd][key]
  __shared__ float Sc[QT * KT];
  __shared__ __attribute__((aligned(32))) bf16_t Pb[QT * KT];
  __shared__ float rowm[QT], rowl[QT], rowa[QT];

  const int tid = threadIdx.x, lane = tid & 31, wave = tid >> 5;
  const int half = lane >> 4, r = lane & 15;
  const int qsub = wave & 3, hh = wave >> 2;
  const int q0 = blockIdx.x * QT;
  const int h  = blockIdx.y;
  const int b  = blockIdx.z;
  const int kvh = h >> 1;  // NH/NKV = 2

  const bf16_t* qbase  = Q + (((long)b * NH_ + h) * S_ + q0) * HD_;
  const bf16_t* kbase  = K + (((long)b * NKV_ + kvh) * S_) * HD_;
  const bf16_t* vtbase = V + ((long)b * NKV_ + kvh) * HD_ * S_;

#pragma unroll
  for (int it = 0; it < 8; it++) {
    int e = it * 2048 + tid * 8;
    async_b128(qbase + e, &Qs[e]);
  }
  if (tid < QT) { rowm[tid] = -1e30f; rowl[tid] = 0.f; }
  v8f o[8];
#pragma unroll
  for (int t = 0; t < 8; t++) o[t] = (v8f){};
  ASYNC_WAIT(0);
  __syncthreads();

  const float scl = 0.0625f;  // 256^-0.5

  for (int k0 = 0; k0 < q0 + QT; k0 += KT) {
    // --- stage K (rows) and Vt (contiguous, already transposed) via async ---
#pragma unroll
    for (int it = 0; it < 4; it++) {
      int e = it * 2048 + tid * 8;
      async_b128(kbase + (long)k0 * HD_ + e, &Ks[e]);
      int hd = e >> 5, kseg = e & 31;
      async_b128(vtbase + (long)hd * S_ + k0 + kseg, &Vt[e]);
    }
    ASYNC_WAIT(0);
    __syncthreads();

    // ---- scores: wave tile rows qsub*16..+15, keys hh*16..+15 ----
    v8f s8 = (v8f){};
#pragma unroll
    for (int d0 = 0; d0 < HD_; d0 += 32) {
      v16bf af = a_frag(&Qs[qsub * 16 * HD_ + d0], r, half, HD_);
      v16bf bfrag = *(const v16bf*)&Ks[(hh * 16 + r) * HD_ + d0 + half * 16];
      s8 = WMMA_BF16(af, bfrag, s8);
    }
#pragma unroll
    for (int i = 0; i < 8; i++) {
      int row = qsub * 16 + i + 8 * half;
      int col = hh * 16 + r;
      float sv = 50.0f * tanhf(s8[i] * scl * (1.0f / 50.0f));  // soft-cap
      if (k0 + col > q0 + row) sv = -1e30f;                    // causal mask
      Sc[row * KT + col] = sv;
    }
    __syncthreads();

    // ---- online softmax stats (one thread per q row) ----
    if (tid < QT) {
      const int row = tid;
      float mo = rowm[row], mx = mo;
#pragma unroll
      for (int c = 0; c < KT; c++) mx = fmaxf(mx, Sc[row * KT + c]);
      float alpha = expf(mo - mx);
      float l = rowl[row] * alpha;
#pragma unroll
      for (int c = 0; c < KT; c++) {
        float p = expf(Sc[row * KT + c] - mx);
        Pb[row * KT + c] = (bf16_t)p;
        l += p;
      }
      rowm[row] = mx; rowl[row] = l; rowa[row] = alpha;
    }
    __syncthreads();

    // ---- rescale O, accumulate P.V over this wave's HD half ----
    float av[8];
#pragma unroll
    for (int i = 0; i < 8; i++) av[i] = rowa[qsub * 16 + i + 8 * half];
#pragma unroll
    for (int t = 0; t < 8; t++)
#pragma unroll
      for (int i = 0; i < 8; i++) o[t][i] *= av[i];

    v16bf pf = a_frag(&Pb[qsub * 16 * KT], r, half, KT);
#pragma unroll
    for (int t = 0; t < 8; t++) {
      int c0 = hh * 128 + t * 16;
      v16bf vf = *(const v16bf*)&Vt[(c0 + r) * KT + half * 16];
      o[t] = WMMA_BF16(pf, vf, o[t]);
    }
    __syncthreads();
  }

  // ---- epilogue: divide by l, write bf16 [B,S,NH*HD] ----
  float li[8];
#pragma unroll
  for (int i = 0; i < 8; i++) li[i] = 1.0f / rowl[qsub * 16 + i + 8 * half];
#pragma unroll
  for (int t = 0; t < 8; t++) {
#pragma unroll
    for (int i = 0; i < 8; i++) {
      int row = qsub * 16 + i + 8 * half;
      int col = hh * 128 + t * 16 + r;
      O[((long)b * S_ + q0 + row) * (NH_ * HD_) + h * HD_ + col] = (bf16_t)(o[t][i] * li[i]);
    }
  }
}

// ---------------------------------------------------------------------------
// GeGLU: out = bf16( gelu_tanh(gate) * up )
// ---------------------------------------------------------------------------
__global__ void k_geglu(const float* __restrict__ gate, const float* __restrict__ up,
                        bf16_t* __restrict__ out, long n) {
  long i = (long)blockIdx.x * blockDim.x + threadIdx.x;
  long stride = (long)gridDim.x * blockDim.x;
  for (; i < n; i += stride) {
    float x = gate[i];
    float g = 0.5f * x * (1.0f + tanhf(0.7978845608f * (x + 0.044715f * x * x * x)));
    out[i] = (bf16_t)(g * up[i]);
  }
}

// ---------------------------------------------------------------------------
// Host orchestration
// ---------------------------------------------------------------------------
extern "C" void kernel_launch(void* const* d_in, const int* in_sizes, int n_in,
                              void* d_out, int out_size, void* d_ws, size_t ws_size,
                              hipStream_t stream) {
  const float* x    = (const float*)d_in[0];
  // d_in[1] = mask (handled analytically via causal logic)
  const float* wq   = (const float*)d_in[2];
  const float* wk   = (const float*)d_in[3];
  const float* wv   = (const float*)d_in[4];
  const float* wo   = (const float*)d_in[5];
  const float* wg   = (const float*)d_in[6];
  const float* wu   = (const float*)d_in[7];
  const float* wd   = (const float*)d_in[8];
  const float* g_in = (const float*)d_in[9];
  const float* g_pa = (const float*)d_in[10];
  const float* g_pf = (const float*)d_in[11];
  const float* g_po = (const float*)d_in[12];
  float* out = (float*)d_out;

  char* ws = (char*)d_ws;
  size_t off = 0;
  auto take = [&](size_t bytes) -> char* {
    char* p = ws + off;
    off += (bytes + 255) & ~(size_t)255;
    return p;
  };

  const size_t QN = (size_t)NH_ * HD_;   // 2048
  const size_t KN = (size_t)NKV_ * HD_;  // 1024

  // bf16 weights, transposed to [N][K]
  bf16_t* wq_b = (bf16_t*)take(2 * (size_t)H_ * QN);
  bf16_t* wk_b = (bf16_t*)take(2 * (size_t)H_ * KN);
  bf16_t* wv_b = (bf16_t*)take(2 * (size_t)H_ * KN);
  bf16_t* wo_b = (bf16_t*)take(2 * QN * (size_t)H_);
  bf16_t* wg_b = (bf16_t*)take(2 * (size_t)H_ * FF_);
  bf16_t* wu_b = (bf16_t*)take(2 * (size_t)H_ * FF_);
  bf16_t* wd_b = (bf16_t*)take(2 * (size_t)FF_ * H_);
  // activations
  bf16_t* h_b    = (bf16_t*)take(2 * (size_t)M_ * H_);
  float*  q_f    = (float*)take(4 * (size_t)M_ * QN);
  float*  k_f    = (float*)take(4 * (size_t)M_ * KN);
  float*  v_f    = (float*)take(4 * (size_t)M_ * KN);
  bf16_t* q_b    = (bf16_t*)take(2 * (size_t)M_ * QN);
  bf16_t* k_b    = (bf16_t*)take(2 * (size_t)M_ * KN);
  bf16_t* v_b    = (bf16_t*)take(2 * (size_t)M_ * KN);
  bf16_t* o_b    = (bf16_t*)take(2 * (size_t)M_ * QN);
  float*  attn_f = (float*)take(4 * (size_t)M_ * H_);
  float*  h2_f   = (float*)take(4 * (size_t)M_ * H_);
  bf16_t* f_b    = (bf16_t*)take(2 * (size_t)M_ * H_);
  float*  gate_f = (float*)take(4 * (size_t)M_ * FF_);
  float*  up_f   = (float*)take(4 * (size_t)M_ * FF_);
  bf16_t* act_b  = (bf16_t*)take(2 * (size_t)M_ * FF_);
  float*  mlp_f  = (float*)take(4 * (size_t)M_ * H_);

  // 1) weights -> bf16 transposed [N][K]
  auto castT = [&](const float* s, bf16_t* d, int K, int N) {
    k_cast_transpose<<<dim3(N / 64, K / 64), 256, 0, stream>>>(s, d, K, N);
  };
  castT(wq, wq_b, H_, (int)QN);
  castT(wk, wk_b, H_, (int)KN);
  castT(wv, wv_b, H_, (int)KN);
  castT(wo, wo_b, (int)QN, H_);
  castT(wg, wg_b, H_, FF_);
  castT(wu, wu_b, H_, FF_);
  castT(wd, wd_b, FF_, H_);

  // 2) h = rmsnorm(x, g_in) -> bf16
  k_rmsnorm<<<M_, 256, 0, stream>>>(x, g_in, nullptr, nullptr, h_b, H_);

  // 3) QKV projections (WMMA)
  k_gemm_bf16<<<dim3(QN / 128, M_ / 128), 256, 0, stream>>>(h_b, wq_b, q_f, M_, (int)QN, H_);
  k_gemm_bf16<<<dim3(KN / 128, M_ / 128), 256, 0, stream>>>(h_b, wk_b, k_f, M_, (int)KN, H_);
  k_gemm_bf16<<<dim3(KN / 128, M_ / 128), 256, 0, stream>>>(h_b, wv_b, v_f, M_, (int)KN, H_);

  // 4) RoPE + attention layouts
  {
    long nq = (long)B_ * S_ * NH_ * (HD_ / 2);
    long nk = (long)B_ * S_ * NKV_ * (HD_ / 2);
    long nv = (long)B_ * S_ * NKV_ * HD_;
    k_rope<<<(int)((nq + 255) / 256), 256, 0, stream>>>(q_f, q_b, NH_);
    k_rope<<<(int)((nk + 255) / 256), 256, 0, stream>>>(k_f, k_b, NKV_);
    k_vperm<<<(int)((nv + 255) / 256), 256, 0, stream>>>(v_f, v_b);
  }

  // 5) flash attention (WMMA QK^T and P.V)
  k_attn<<<dim3(S_ / 64, NH_, B_), 256, 0, stream>>>(q_b, k_b, v_b, o_b);

  // 6) O-projection
  k_gemm_bf16<<<dim3(H_ / 128, M_ / 128), 256, 0, stream>>>(o_b, wo_b, attn_f, M_, H_, (int)QN);

  // 7) h2 = x + rmsnorm(attn_out, g_post_attn); f = rmsnorm(h2, g_pre_ff) bf16
  k_rmsnorm<<<M_, 256, 0, stream>>>(attn_f, g_pa, x, h2_f, nullptr, H_);
  k_rmsnorm<<<M_, 256, 0, stream>>>(h2_f, g_pf, nullptr, nullptr, f_b, H_);

  // 8) GeGLU MLP
  k_gemm_bf16<<<dim3(FF_ / 128, M_ / 128), 256, 0, stream>>>(f_b, wg_b, gate_f, M_, FF_, H_);
  k_gemm_bf16<<<dim3(FF_ / 128, M_ / 128), 256, 0, stream>>>(f_b, wu_b, up_f, M_, FF_, H_);
  k_geglu<<<4096, 256, 0, stream>>>(gate_f, up_f, act_b, (long)M_ * FF_);
  k_gemm_bf16<<<dim3(H_ / 128, M_ / 128), 256, 0, stream>>>(act_b, wd_b, mlp_f, M_, H_, FF_);

  // 9) out = h2 + rmsnorm(mlp, g_post_ff)
  k_rmsnorm<<<M_, 256, 0, stream>>>(mlp_f, g_po, h2_f, out, nullptr, H_);
}
